// TransformerSampling_41893111005466
// MI455X (gfx1250) — matrix operand order repaired and checked by reading, not compiled, not run on hardware
//
#include <hip/hip_runtime.h>
#include <hip/hip_bf16.h>

typedef __attribute__((ext_vector_type(8)))  float  v8f;
typedef __attribute__((ext_vector_type(16))) __bf16 v16bf;
typedef __attribute__((ext_vector_type(8)))  __bf16 v8bf;

#define DEV __device__ __forceinline__

// ---------------------------------------------------------------------------
// WMMA fragment loaders (wave32; A/B share a contiguous k order, not the table order of CDNA5 ISA 7.12.2)
// A 16x32 bf16: lane row = lane&15; K = (lane>>4)*16 + [0..15]
//               (two contiguous 16B loads)
// B 32x16 bf16: lane col = lane&15; K = (lane>>4)*16 + [0..15] (one 32B load)
// C/D f32:      VGPR r -> row r + 8*(lane>>4), col = lane&15
// ---------------------------------------------------------------------------
DEV v16bf ldA(const __bf16* p, int lane) {
    p += ((lane >> 4) << 4);
    v8bf lo = *(const v8bf*)p;
    v8bf hi = *(const v8bf*)(p + 8);
    return __builtin_shufflevector(lo, hi, 0,1,2,3,4,5,6,7,8,9,10,11,12,13,14,15);
}
DEV v16bf ldB(const __bf16* p, int lane) {
    return *(const v16bf*)(p + ((lane >> 4) << 4));
}
DEV v8f wmma_bf16(v16bf a, v16bf b, v8f c) {
    return __builtin_amdgcn_wmma_f32_16x16x32_bf16(false, a, false, b, (short)0, c, false, false);
}
DEV int region16(int g) { return g < 240 ? 0 : (g < 248 ? 1 : 2); }

// ---------------------------------------------------------------------------
// Constants: B=2, C=96, H=W=128, heads=3, hd=32, WS=8, WS_UP=16
// ---------------------------------------------------------------------------

// K0: f32 -> bf16 weight conversion
__global__ __launch_bounds__(256) void k_cvt(const float* __restrict__ s,
                                             __bf16* __restrict__ d, int n) {
    int i = blockIdx.x * 256 + threadIdx.x;
    if (i < n) d[i] = (__bf16)s[i];
}

// K1: LayerNorm + roll(-4,-4) + window partition, write x^T per window:
//     xwT[b_][c][n]  (512 windows, 96 x 64, bf16)  -- A operand of qkv GEMM
__global__ __launch_bounds__(256) void k_prep(const float* __restrict__ x,
                                              const float* __restrict__ g,
                                              const float* __restrict__ be,
                                              __bf16* __restrict__ xwT) {
    int id = blockIdx.x * 256 + threadIdx.x;          // 512*64 = 32768
    int b_ = id >> 6, n = id & 63;
    int b = b_ >> 8, wIdx = b_ & 255;
    int sy = (((wIdx >> 4) << 3) + (n >> 3) + 4) & 127;
    int sx = (((wIdx & 15) << 3) + (n & 7) + 4) & 127;
    const float* xb = x + (size_t)b * 96 * 16384 + sy * 128 + sx;
    float mu = 0.f;
    for (int c = 0; c < 96; c++) mu += xb[c * 16384];
    mu *= (1.f / 96.f);
    float var = 0.f;
    for (int c = 0; c < 96; c++) { float d = xb[c * 16384] - mu; var += d * d; }
    var *= (1.f / 96.f);
    float rstd = rsqrtf(var + 1e-5f);
    for (int c = 0; c < 96; c++)
        xwT[((size_t)b_ * 96 + c) * 64 + n] =
            (__bf16)((xb[c * 16384] - mu) * rstd * g[c] + be[c]);
}

// K2: per-window qkv GEMM: (96x64) @ (64x768) + bias; scatter into
//     q[b,h,n,d]*scale, k[b,h,n,d], v[b,h,d,n]  (bf16)
__global__ __launch_bounds__(256) void k_qkv(const __bf16* __restrict__ xwT,
                                             const __bf16* __restrict__ wq,
                                             const float* __restrict__ qb_,
                                             __bf16* __restrict__ q,
                                             __bf16* __restrict__ k,
                                             __bf16* __restrict__ v) {
    int lane = threadIdx.x & 31, wave = threadIdx.x >> 5;
    int tile = blockIdx.x * 8 + wave;                 // 512*288 = 147456 tiles
    int w = tile / 288, r = tile % 288;
    int m0 = (r / 48) * 16, n0 = (r % 48) * 16;
    const __bf16* A = xwT + (size_t)w * 96 * 64;
    v8f acc = {0.f,0.f,0.f,0.f,0.f,0.f,0.f,0.f};
    for (int k0 = 0; k0 < 64; k0 += 32) {
        v16bf a = ldA(A + (m0 + (lane & 15)) * 64 + k0, lane);
        v16bf b = ldB(wq + (n0 + (lane & 15)) * 64 + k0, lane);
        acc = wmma_bf16(a, b, acc);
    }
    int col = n0 + (lane & 15);
    int tok = col & 255;
    // `which` is wave-uniform (tiles never straddle a 256-col boundary):
    // force it scalar so the 3-way scatter is a uniform branch, not exec-mask.
    int which = __builtin_amdgcn_readfirstlane(col >> 8);
    float bias = qb_[col];
    const float scale = 0.17677669529663687f;         // 1/sqrt(32)
    if (which == 0) {
#pragma unroll
        for (int rr = 0; rr < 8; rr++) {
            int c = m0 + rr + ((lane >> 4) << 3);
            size_t bh = (size_t)(w * 3 + (c >> 5));
            q[(bh * 256 + tok) * 32 + (c & 31)] = (__bf16)((acc[rr] + bias) * scale);
        }
    } else if (which == 1) {
#pragma unroll
        for (int rr = 0; rr < 8; rr++) {
            int c = m0 + rr + ((lane >> 4) << 3);
            size_t bh = (size_t)(w * 3 + (c >> 5));
            k[(bh * 256 + tok) * 32 + (c & 31)] = (__bf16)(acc[rr] + bias);
        }
    } else {
#pragma unroll
        for (int rr = 0; rr < 8; rr++) {
            int c = m0 + rr + ((lane >> 4) << 3);
            size_t bh = (size_t)(w * 3 + (c >> 5));
            v[(bh * 32 + (c & 31)) * 256 + tok] = (__bf16)(acc[rr] + bias);
        }
    }
}

// K3: attention, one block per (window,head); 16 waves x 16-row strips.
//     Flash-style over 32-column chunks: S=WMMA(q,k^T)+bias+mask -> LDS ->
//     online softmax -> P (bf16, A-layout via LDS) -> O += WMMA(P, V).
__global__ __launch_bounds__(512) void k_attn(const __bf16* __restrict__ q,
                                              const __bf16* __restrict__ k,
                                              const __bf16* __restrict__ v,
                                              const float* __restrict__ rel_t,
                                              __bf16* __restrict__ y) {
    __shared__ float  sS[16][16 * 36];   // padded rows (36 f32) vs bank conflicts
    __shared__ __bf16 sP[16][16 * 32];   // 16B-aligned rows for A-frag loads
    __shared__ float  sAl[16][16];
    __shared__ float  sL[16][16];

    const int lane = threadIdx.x & 31;
    const int wave = threadIdx.x >> 5;   // strip id (16 rows each)
    const int hl = lane >> 4, ll = lane & 15;
    const int bh = blockIdx.x;           // 0..1535
    const int b_ = bh / 3, h = bh - b_ * 3;
    const int wIdx = b_ & 255;
    const int wy16 = (wIdx >> 4) << 4, wx16 = (wIdx & 15) << 4;

    const __bf16* qbp = q + (size_t)bh * 256 * 32;
    const __bf16* kbp = k + (size_t)bh * 256 * 32;
    const __bf16* vbp = v + (size_t)bh * 32 * 256;

    v16bf aq = ldA(qbp + (wave * 16 + ll) * 32, lane);   // K = full hd = 32

    // loop-invariant per-row constants (bias row term, shift-mask region)
    int rowrel[8], regi[8];
#pragma unroll
    for (int rr = 0; rr < 8; rr++) {
        int i = wave * 16 + rr + (hl << 3);
        int iy = i >> 4, ix = i & 15;
        rowrel[rr] = ((iy + 15) * 31 + (ix + 15)) * 3 + h;
        regi[rr]   = region16(wy16 + iy) * 3 + region16(wx16 + ix);
    }

    v8f o0 = {0.f,0.f,0.f,0.f,0.f,0.f,0.f,0.f};
    v8f o1 = {0.f,0.f,0.f,0.f,0.f,0.f,0.f,0.f};
    float m_run = -3.4e38f, l_run = 0.f;                 // row = ll

    for (int j0 = 0; j0 < 256; j0 += 32) {
        if (j0 + 32 < 256) {  // pull next chunk toward the WGP (global_prefetch_b8)
            __builtin_prefetch((const void*)(kbp + (j0 + 32 + ll) * 32), 0, 1);
            __builtin_prefetch((const void*)(vbp + ll * 256 + j0 + 32), 0, 1);
        }
        v16bf bk0 = ldB(kbp + (j0 + ll) * 32, lane);
        v16bf bk1 = ldB(kbp + (j0 + 16 + ll) * 32, lane);
        v8f z = {0.f,0.f,0.f,0.f,0.f,0.f,0.f,0.f};
        v8f s0 = wmma_bf16(aq, bk0, z);
        v8f s1 = wmma_bf16(aq, bk1, z);

        int jA = j0 + ll, jB = jA + 16;
        int jAy = jA >> 4, jAx = jA & 15, jBy = jB >> 4, jBx = jB & 15;
        int colA = (jAy * 31 + jAx) * 3, colB = (jBy * 31 + jBx) * 3;
        int regA = region16(wy16 + jAy) * 3 + region16(wx16 + jAx);
        int regB = region16(wy16 + jBy) * 3 + region16(wx16 + jBx);
#pragma unroll
        for (int rr = 0; rr < 8; rr++) {
            int rloc = rr + (hl << 3);
            float bA = rel_t[rowrel[rr] - colA];
            float bB = rel_t[rowrel[rr] - colB];
            sS[wave][rloc * 36 + ll]      = s0[rr] + bA + (regi[rr] != regA ? -100.f : 0.f);
            sS[wave][rloc * 36 + 16 + ll] = s1[rr] + bB + (regi[rr] != regB ? -100.f : 0.f);
        }
        asm volatile("s_wait_dscnt 0x0" ::: "memory");

        // online softmax for row ll (upper half-wave duplicates harmlessly)
        float mloc = -3.4e38f;
        for (int c = 0; c < 32; c++) mloc = fmaxf(mloc, sS[wave][ll * 36 + c]);
        float mnew = fmaxf(m_run, mloc);
        float alpha = expf(m_run - mnew);
        float lsum = 0.f;
        for (int c = 0; c < 32; c++) {
            float p = expf(sS[wave][ll * 36 + c] - mnew);
            lsum += p;
            sP[wave][ll * 32 + c] = (__bf16)p;
        }
        l_run = l_run * alpha + lsum;
        m_run = mnew;
        sAl[wave][ll] = alpha;
        asm volatile("s_wait_dscnt 0x0" ::: "memory");

#pragma unroll
        for (int rr = 0; rr < 8; rr++) {
            float al = sAl[wave][rr + (hl << 3)];
            o0[rr] *= al; o1[rr] *= al;
        }
        v16bf pf  = ldA(&sP[wave][ll * 32], lane);             // P in A-layout
        v16bf bv0 = ldB(vbp + ll * 256 + j0, lane);            // d = ll
        v16bf bv1 = ldB(vbp + (16 + ll) * 256 + j0, lane);     // d = 16+ll
        o0 = wmma_bf16(pf, bv0, o0);
        o1 = wmma_bf16(pf, bv1, o1);
    }
    sL[wave][ll] = l_run;
    asm volatile("s_wait_dscnt 0x0" ::: "memory");
#pragma unroll
    for (int rr = 0; rr < 8; rr++) {
        int rloc = rr + (hl << 3);
        float inv = 1.f / sL[wave][rloc];
        size_t t = (size_t)b_ * 256 + wave * 16 + rloc;
        y[t * 96 + h * 32 + ll]      = (__bf16)(o0[rr] * inv);
        y[t * 96 + h * 32 + 16 + ll] = (__bf16)(o1[rr] * inv);
    }
}

// K4: proj GEMM (131072x96 @ 96x96) fused with window_reverse + roll(+8,+8)
//     + bilinear 2x upsampled residual -> xo (B,256,256,96) f32
__global__ __launch_bounds__(256) void k_proj(const __bf16* __restrict__ y,
                                              const __bf16* __restrict__ pw,
                                              const float* __restrict__ pb,
                                              const float* __restrict__ x,
                                              float* __restrict__ xo) {
    int lane = threadIdx.x & 31, wave = threadIdx.x >> 5;
    int tile = blockIdx.x * 8 + wave;                 // 8192*6 = 49152 tiles
    int s = tile / 6, n0 = (tile % 6) * 16;
    const __bf16* A = y + (size_t)s * 16 * 96;
    v8f acc = {0.f,0.f,0.f,0.f,0.f,0.f,0.f,0.f};
    for (int k0 = 0; k0 < 96; k0 += 32) {
        v16bf a = ldA(A + (lane & 15) * 96 + k0, lane);
        v16bf b = ldB(pw + (n0 + (lane & 15)) * 96 + k0, lane);
        acc = wmma_bf16(a, b, acc);
    }
    int c = n0 + (lane & 15);
    float bias = pb[c];
#pragma unroll
    for (int rr = 0; rr < 8; rr++) {
        int t = s * 16 + rr + ((lane >> 4) << 3);
        int b_ = t >> 8, n = t & 255;
        int b = b_ >> 8, wIdx = b_ & 255;
        int gy = ((wIdx >> 4) << 4) + (n >> 4);
        int gx = ((wIdx & 15) << 4) + (n & 15);
        int fy = (gy + 8) & 255, fx = (gx + 8) & 255;
        // bilinear (half-pixel, edge clamp) of x (B,C,128,128) at (fy,fx)
        float iyf = fy * 0.5f - 0.25f;
        int i0 = (int)floorf(iyf); float fyf = iyf - (float)i0;
        float ixf = fx * 0.5f - 0.25f;
        int j0 = (int)floorf(ixf); float fxf = ixf - (float)j0;
        int i0a = i0 < 0 ? 0 : i0, i1a = i0 + 1 > 127 ? 127 : i0 + 1;
        int j0a = j0 < 0 ? 0 : j0, j1a = j0 + 1 > 127 ? 127 : j0 + 1;
        const float* xb = x + ((size_t)(b * 96 + c)) * 16384;
        float sc = (1.f - fyf) * ((1.f - fxf) * xb[i0a * 128 + j0a] + fxf * xb[i0a * 128 + j1a])
                 + fyf         * ((1.f - fxf) * xb[i1a * 128 + j0a] + fxf * xb[i1a * 128 + j1a]);
        xo[((size_t)(b * 65536) + fy * 256 + fx) * 96 + c] = acc[rr] + bias + sc;
    }
}

// K5: second LayerNorm over xo rows -> ln bf16
__global__ __launch_bounds__(256) void k_ln2(const float* __restrict__ xo,
                                             const float* __restrict__ g,
                                             const float* __restrict__ be,
                                             __bf16* __restrict__ ln) {
    int t = blockIdx.x * 256 + threadIdx.x;           // 131072
    const float* r = xo + (size_t)t * 96;
    float mu = 0.f;
    for (int c = 0; c < 96; c++) mu += r[c];
    mu *= (1.f / 96.f);
    float var = 0.f;
    for (int c = 0; c < 96; c++) { float d = r[c] - mu; var += d * d; }
    var *= (1.f / 96.f);
    float rstd = rsqrtf(var + 1e-5f);
    for (int c = 0; c < 96; c++)
        ln[(size_t)t * 96 + c] = (__bf16)((r[c] - mu) * rstd * g[c] + be[c]);
}

// K6: fused MLP per 16-token strip: fc1(96->384)+exact GELU staged in LDS,
//     fc2(384->96) + residual -> final output. 6 waves per block.
__global__ __launch_bounds__(192) void k_mlp(const __bf16* __restrict__ ln,
                                             const __bf16* __restrict__ w1,
                                             const float* __restrict__ b1,
                                             const __bf16* __restrict__ w2,
                                             const float* __restrict__ b2,
                                             const float* __restrict__ xo,
                                             float* __restrict__ out) {
    __shared__ __bf16 h1s[16 * 384];
    int lane = threadIdx.x & 31, wave = threadIdx.x >> 5;  // 0..5
    int s = blockIdx.x;                                     // 8192 strips
    const __bf16* A = ln + (size_t)s * 16 * 96;
    for (int nt = wave * 4; nt < wave * 4 + 4; nt++) {
        int n0 = nt * 16;
        v8f acc = {0.f,0.f,0.f,0.f,0.f,0.f,0.f,0.f};
        for (int k0 = 0; k0 < 96; k0 += 32) {
            v16bf a = ldA(A + (lane & 15) * 96 + k0, lane);
            v16bf b = ldB(w1 + (n0 + (lane & 15)) * 96 + k0, lane);
            acc = wmma_bf16(a, b, acc);
        }
        int c = n0 + (lane & 15);
        float bb = b1[c];
#pragma unroll
        for (int rr = 0; rr < 8; rr++) {
            int row = rr + ((lane >> 4) << 3);
            float z = acc[rr] + bb;
            float gl = 0.5f * z * (1.0f + erff(z * 0.70710678118654752f));
            h1s[row * 384 + c] = (__bf16)gl;
        }
    }
    __syncthreads();
    {
        int n0 = wave * 16;
        v8f acc = {0.f,0.f,0.f,0.f,0.f,0.f,0.f,0.f};
        for (int k0 = 0; k0 < 384; k0 += 32) {
            v16bf a = ldA(h1s + (lane & 15) * 384 + k0, lane);
            v16bf b = ldB(w2 + (n0 + (lane & 15)) * 384 + k0, lane);
            acc = wmma_bf16(a, b, acc);
        }
        int c = n0 + (lane & 15);
        float bb = b2[c];
#pragma unroll
        for (int rr = 0; rr < 8; rr++) {
            size_t t = (size_t)s * 16 + rr + ((lane >> 4) << 3);
            out[t * 96 + c] = acc[rr] + bb + xo[t * 96 + c];
        }
    }
}

// ---------------------------------------------------------------------------
extern "C" void kernel_launch(void* const* d_in, const int* in_sizes, int n_in,
                              void* d_out, int out_size, void* d_ws, size_t ws_size,
                              hipStream_t stream) {
    const float* x      = (const float*)d_in[0];
    const float* norm_g = (const float*)d_in[1];
    const float* norm_b = (const float*)d_in[2];
    const float* qkv_w  = (const float*)d_in[3];
    const float* qkv_b  = (const float*)d_in[4];
    const float* rel_t  = (const float*)d_in[5];
    const float* proj_w = (const float*)d_in[6];
    const float* proj_b = (const float*)d_in[7];
    const float* fc1_w  = (const float*)d_in[8];
    const float* fc1_b  = (const float*)d_in[9];
    const float* fc2_w  = (const float*)d_in[10];
    const float* fc2_b  = (const float*)d_in[11];
    float* out = (float*)d_out;
    char* w = (char*)d_ws;

    size_t off = 0;
    auto take = [&](size_t bytes) {
        size_t o = off; off = (off + bytes + 255) & ~(size_t)255; return o;
    };
    __bf16* qkvw_h = (__bf16*)(w + take((size_t)49152 * 2));
    __bf16* projw_h = (__bf16*)(w + take((size_t)9216 * 2));
    __bf16* fc1w_h = (__bf16*)(w + take((size_t)36864 * 2));
    __bf16* fc2w_h = (__bf16*)(w + take((size_t)36864 * 2));
    __bf16* xwT = (__bf16*)(w + take((size_t)512 * 96 * 64 * 2));
    __bf16* qB  = (__bf16*)(w + take((size_t)512 * 3 * 256 * 32 * 2));
    __bf16* kB  = (__bf16*)(w + take((size_t)512 * 3 * 256 * 32 * 2));
    __bf16* vB  = (__bf16*)(w + take((size_t)512 * 3 * 32 * 256 * 2));
    __bf16* yB  = (__bf16*)(w + take((size_t)512 * 256 * 96 * 2));
    float*  xo  = (float*)(w + take((size_t)2 * 256 * 256 * 96 * 4));
    __bf16* lnB = (__bf16*)(w + take((size_t)131072 * 96 * 2));
    (void)ws_size; (void)in_sizes; (void)n_in; (void)out_size;

    k_cvt<<<(49152 + 255) / 256, 256, 0, stream>>>(qkv_w, qkvw_h, 49152);
    k_cvt<<<(9216  + 255) / 256, 256, 0, stream>>>(proj_w, projw_h, 9216);
    k_cvt<<<(36864 + 255) / 256, 256, 0, stream>>>(fc1_w, fc1w_h, 36864);
    k_cvt<<<(36864 + 255) / 256, 256, 0, stream>>>(fc2_w, fc2w_h, 36864);

    k_prep<<<128, 256, 0, stream>>>(x, norm_g, norm_b, xwT);
    k_qkv<<<18432, 256, 0, stream>>>(xwT, qkvw_h, qkv_b, qB, kB, vB);
    k_attn<<<1536, 512, 0, stream>>>(qB, kB, vB, rel_t, yB);
    k_proj<<<6144, 256, 0, stream>>>(yB, projw_h, proj_b, x, xo);
    k_ln2<<<512, 256, 0, stream>>>(xo, norm_g, norm_b, lnB);
    k_mlp<<<8192, 192, 0, stream>>>(lnB, fc1w_h, fc1_b, fc2w_h, fc2_b, xo, out);
}
